// Multi_Head_Attention_23270132810445
// MI455X (gfx1250) — matrix operand order repaired and checked by reading, not compiled, not run on hardware
//
#include <hip/hip_runtime.h>

// ---------------------------------------------------------------------------
// MHA for MI455X (gfx1250): bf16 WMMA (v_wmma_f32_16x16x32_bf16), f32 accum.
// Pipeline: 3x projection GEMMs -> per-head attention -> output GEMM.
// Native __bf16 storage so f32->bf16 uses v_cvt_pk_bf16_f32.
// Q-strip staging uses async global->LDS copies (ASYNCcnt) when available.
// ---------------------------------------------------------------------------

typedef __attribute__((ext_vector_type(16))) __bf16 v16bf;
typedef __attribute__((ext_vector_type(8)))  __bf16 v8bf;
typedef __attribute__((ext_vector_type(8)))  float  v8f;
typedef int v4i_gcc __attribute__((vector_size(16)));   // matches builtin sig

#define DM          256
#define NH          8
#define NQK         2048        // DM*NH
#define ROWS        16384       // B*D1*D2
#define HEAD_ELEMS  65536       // D2*DM
#define NHEADS      512         // B*D1*NH
#define HEAD_BUF    ((size_t)NHEADS * HEAD_ELEMS)   // 33.55M bf16 elems = 64MB

#if defined(__has_builtin)
#if __has_builtin(__builtin_amdgcn_global_load_async_to_lds_b128) && \
    __has_builtin(__builtin_amdgcn_s_wait_asynccnt)
#define HAVE_ASYNC_LDS 1
#endif
#endif
#ifndef HAVE_ASYNC_LDS
#define HAVE_ASYNC_LDS 0
#endif

union Frag16 { v16bf v; v8bf h[2]; };

// ---------------------------------------------------------------------------
// Projection GEMM: Y = X(16384x256) * W(256x2048) + b, stored bf16,
// de-interleaved per head.  transposed==0 -> dst[head][pos][d] (Q, K)
//                           transposed==1 -> dst[head][d][pos] (V, for O=P*V)
// Block: 256 thr (8 waves, 2x4 grid), tile 64(M) x 256(N), K-step 32.
// Each wave: 2x4 = 8 WMMA tiles (32x64 of output), 6 fragment loads / step.
// ---------------------------------------------------------------------------
__global__ __launch_bounds__(256) void proj_kernel(
    const float* __restrict__ x, const float* __restrict__ W,
    const float* __restrict__ bias, __bf16* __restrict__ dst,
    int transposed)
{
    __shared__ alignas(16) __bf16 Asl[64][32];    // [m][k]
    __shared__ alignas(16) __bf16 Bsl[256][32];   // B^T: [n][k]

    const int tid   = threadIdx.x;
    const int lane  = tid & 31;
    const int w     = tid >> 5;
    const int laneN = lane & 15;
    const int laneG = lane >> 4;
    const int wm = w & 1, wn = w >> 1;            // 2 x 4 wave grid
    const int mbase = blockIdx.y * 64;
    const int nbase = blockIdx.x * 256;

    v8f acc[2][4];
    #pragma unroll
    for (int tm = 0; tm < 2; ++tm)
      #pragma unroll
      for (int tn = 0; tn < 4; ++tn) {
        float bv = bias[nbase + wn*64 + tn*16 + laneN];
        #pragma unroll
        for (int v = 0; v < 8; ++v) acc[tm][tn][v] = bv;
      }

    const int ai = tid >> 2;          // 0..63  (A stage row)
    const int ak = (tid & 3) * 8;     // A stage k chunk

    for (int kb = 0; kb < DM; kb += 32) {
        #pragma unroll
        for (int e = 0; e < 8; ++e)
            Asl[ai][ak + e] = (__bf16)x[(mbase + ai) * DM + kb + ak + e];
        #pragma unroll
        for (int e = 0; e < 32; ++e)
            Bsl[tid][e] = (__bf16)W[(kb + e) * NQK + nbase + tid];
        if (kb + 32 < DM) {   // prefetch next K step (global_prefetch_b8)
            __builtin_prefetch(&x[(mbase + ai) * DM + kb + 32 + ak], 0, 1);
            __builtin_prefetch(&W[(kb + 32) * NQK + nbase + tid], 0, 1);
        }
        __syncthreads();

        Frag16 aF[2], bF[4];
        #pragma unroll
        for (int tm = 0; tm < 2; ++tm) {
            const __bf16* row = &Asl[wm*32 + tm*16 + laneN][0];
            aF[tm].h[0] = *(const v8bf*)(row + 8*laneG);        // K = 8g..8g+7
            aF[tm].h[1] = *(const v8bf*)(row + 16 + 8*laneG);   // K = 16+8g..
        }
        #pragma unroll
        for (int tn = 0; tn < 4; ++tn) {
            const __bf16* row = &Bsl[wn*64 + tn*16 + laneN][0];
            bF[tn].h[0] = *(const v8bf*)(row + 8*laneG);        // K = 8g..8g+7
            bF[tn].h[1] = *(const v8bf*)(row + 16 + 8*laneG);   // K = 16+8g..
        }
        #pragma unroll
        for (int tm = 0; tm < 2; ++tm)
          #pragma unroll
          for (int tn = 0; tn < 4; ++tn)
            acc[tm][tn] = __builtin_amdgcn_wmma_f32_16x16x32_bf16(
                false, aF[tm].v, false, bF[tn].v, (short)0, acc[tm][tn], false, false);
        __syncthreads();
    }

    // scatter-store per head: col j -> (d = j>>3, head = j&7)
    #pragma unroll
    for (int tm = 0; tm < 2; ++tm)
      #pragma unroll
      for (int tn = 0; tn < 4; ++tn) {
        int col  = nbase + wn*64 + tn*16 + laneN;
        int head = col & 7, dd = col >> 3;
        #pragma unroll
        for (int v = 0; v < 8; ++v) {
            int row = mbase + wm*32 + tm*16 + 8*laneG + v;
            int bc = row >> 8, pos = row & 255;
            size_t base = (size_t)(bc * NH + head) * HEAD_ELEMS;
            size_t addr = transposed ? base + (size_t)dd * 256 + pos
                                     : base + (size_t)pos * 256 + dd;
            dst[addr] = (__bf16)acc[tm][tn][v];
        }
      }
}

// ---------------------------------------------------------------------------
// Attention: one (b,d1,head) problem per blockIdx.y; blockIdx.x = 64-row
// query block; 4 waves, 16 query rows each.  S kept f32 in LDS, softmax via
// cross-lane shfl, P (unnormalized exp) overlays the dead Q strip as bf16.
// ---------------------------------------------------------------------------
__global__ __launch_bounds__(128) void attn_kernel(
    const __bf16* __restrict__ qh,
    const __bf16* __restrict__ kh,
    const __bf16* __restrict__ vT,
    __bf16* __restrict__ oh)
{
    __shared__ alignas(16) __bf16 Qs[4][16][256];  // 32KB: Q strip, then P
    __shared__ alignas(16) float  Ss[4][16][256];  // 64KB: scores f32
    __shared__ float rowsum[4][16];

    const int tid   = threadIdx.x;
    const int lane  = tid & 31;
    const int w     = tid >> 5;            // wave 0..3
    const int laneN = lane & 15;
    const int laneG = lane >> 4;
    const int head  = blockIdx.y;
    const int qrow0 = blockIdx.x * 64 + w * 16;
    const size_t hb = (size_t)head * HEAD_ELEMS;

    // stage this wave's 16 Q rows into LDS (16B chunks)
#if HAVE_ASYNC_LDS
    for (int c = lane; c < 16 * 32; c += 32) {
        int r = c >> 5, off = (c & 31) * 8;
        __builtin_amdgcn_global_load_async_to_lds_b128(
            (__attribute__((address_space(1))) v4i_gcc*)
                (qh + hb + (size_t)(qrow0 + r) * 256 + off),
            (__attribute__((address_space(3))) v4i_gcc*)&Qs[w][r][off],
            0, 0);
    }
    __builtin_amdgcn_s_wait_asynccnt(0);   // LDS visible before ds_load frags
#else
    for (int c = lane; c < 16 * 32; c += 32) {
        int r = c >> 5, off = (c & 31) * 8;
        *(uint4*)&Qs[w][r][off] =
            *(const uint4*)(qh + hb + (size_t)(qrow0 + r) * 256 + off);
    }
#endif

    // S = Q K^T / sqrt(256);  B operand streamed from K rows (contiguous d)
    for (int t = 0; t < 16; ++t) {
        v8f acc = {};
        const __bf16* krow = kh + hb + (size_t)(t*16 + laneN) * 256 + 8*laneG;
        #pragma unroll
        for (int kk = 0; kk < 8; ++kk) {
            int d0 = kk * 32;
            Frag16 a, b;
            a.h[0] = *(const v8bf*)&Qs[w][laneN][d0 + 8*laneG];
            a.h[1] = *(const v8bf*)&Qs[w][laneN][d0 + 16 + 8*laneG];
            b.h[0] = *(const v8bf*)(krow + d0);
            b.h[1] = *(const v8bf*)(krow + d0 + 16);
            acc = __builtin_amdgcn_wmma_f32_16x16x32_bf16(
                false, a.v, false, b.v, (short)0, acc, false, false);
        }
        #pragma unroll
        for (int v = 0; v < 8; ++v)
            Ss[w][8*laneG + v][t*16 + laneN] = acc[v] * 0.0625f;
    }

    // softmax over 256 keys; lanes (r, r+16) split columns, combine via shfl
    {
        int r  = lane & 15;
        int c0 = (lane >> 4) * 128;
        float m = -3.4e38f;
        for (int c = 0; c < 128; ++c) m = fmaxf(m, Ss[w][r][c0 + c]);
        m = fmaxf(m, __shfl_xor(m, 16, 32));
        float s = 0.f;
        for (int c = 0; c < 128; ++c) {
            float e = __expf(Ss[w][r][c0 + c] - m);
            s += e;
            Qs[w][r][c0 + c] = (__bf16)e;   // P (unnormalized)
        }
        s += __shfl_xor(s, 16, 32);
        if (laneG == 0) rowsum[w][r] = s;
    }

    // O = P V  (vT[d][j] gives B^T rows contiguous in j); normalize at store
    for (int t2 = 0; t2 < 16; ++t2) {
        v8f acc = {};
        const __bf16* vrow = vT + hb + (size_t)(t2*16 + laneN) * 256 + 8*laneG;
        #pragma unroll
        for (int kk = 0; kk < 8; ++kk) {
            int j0 = kk * 32;
            Frag16 a, b;
            a.h[0] = *(const v8bf*)&Qs[w][laneN][j0 + 8*laneG];
            a.h[1] = *(const v8bf*)&Qs[w][laneN][j0 + 16 + 8*laneG];
            b.h[0] = *(const v8bf*)(vrow + j0);
            b.h[1] = *(const v8bf*)(vrow + j0 + 16);
            acc = __builtin_amdgcn_wmma_f32_16x16x32_bf16(
                false, a.v, false, b.v, (short)0, acc, false, false);
        }
        #pragma unroll
        for (int v = 0; v < 8; ++v) {
            int rs = 8*laneG + v;
            float inv = 1.0f / rowsum[w][rs];
            oh[hb + (size_t)(qrow0 + rs) * 256 + t2*16 + laneN] =
                (__bf16)(acc[v] * inv);
        }
    }
}

// ---------------------------------------------------------------------------
// Output GEMM: out = O_flat(16384x2048, gathered j=d*8+h) * Wo(2048x256) + bo
// Block tile 64(M) x 256(N=full), K-step 32, 64 K iterations.
// ---------------------------------------------------------------------------
__global__ __launch_bounds__(256) void outproj_kernel(
    const __bf16* __restrict__ oh,
    const float* __restrict__ Wo,
    const float* __restrict__ bo,
    float* __restrict__ out)
{
    __shared__ alignas(16) __bf16 Asl[64][32];
    __shared__ alignas(16) __bf16 Bsl[256][32];

    const int tid   = threadIdx.x;
    const int lane  = tid & 31;
    const int w     = tid >> 5;
    const int laneN = lane & 15;
    const int laneG = lane >> 4;
    const int wm = w & 1, wn = w >> 1;
    const int mbase = blockIdx.y * 64;

    v8f acc[2][4];
    #pragma unroll
    for (int tm = 0; tm < 2; ++tm)
      #pragma unroll
      for (int tn = 0; tn < 4; ++tn) {
        float bv = bo[wn*64 + tn*16 + laneN];
        #pragma unroll
        for (int v = 0; v < 8; ++v) acc[tm][tn][v] = bv;
      }

    const int ai = tid >> 2;
    const int ak = (tid & 3) * 8;

    for (int kb = 0; kb < NQK; kb += 32) {
        const int row = mbase + ai;
        const int bc = row >> 8, pos = row & 255;
        #pragma unroll
        for (int e = 0; e < 8; ++e) {
            int j = kb + ak + e;                       // interleaved column
            Asl[ai][ak + e] =
                oh[(size_t)(bc * NH + (j & 7)) * HEAD_ELEMS + (size_t)pos * 256 + (j >> 3)];
        }
        #pragma unroll
        for (int e = 0; e < 32; ++e)
            Bsl[tid][e] = (__bf16)Wo[(kb + e) * DM + tid];
        if (kb + 32 < NQK)
            __builtin_prefetch(&Wo[(kb + 32) * DM + tid], 0, 1);
        __syncthreads();

        Frag16 aF[2], bF[4];
        #pragma unroll
        for (int tm = 0; tm < 2; ++tm) {
            const __bf16* r = &Asl[wm*32 + tm*16 + laneN][0];
            aF[tm].h[0] = *(const v8bf*)(r + 8*laneG);
            aF[tm].h[1] = *(const v8bf*)(r + 16 + 8*laneG);
        }
        #pragma unroll
        for (int tn = 0; tn < 4; ++tn) {
            const __bf16* r = &Bsl[wn*64 + tn*16 + laneN][0];
            bF[tn].h[0] = *(const v8bf*)(r + 8*laneG);
            bF[tn].h[1] = *(const v8bf*)(r + 16 + 8*laneG);
        }
        #pragma unroll
        for (int tm = 0; tm < 2; ++tm)
          #pragma unroll
          for (int tn = 0; tn < 4; ++tn)
            acc[tm][tn] = __builtin_amdgcn_wmma_f32_16x16x32_bf16(
                false, aF[tm].v, false, bF[tn].v, (short)0, acc[tm][tn], false, false);
        __syncthreads();
    }

    #pragma unroll
    for (int tm = 0; tm < 2; ++tm)
      #pragma unroll
      for (int tn = 0; tn < 4; ++tn) {
        int col = wn*64 + tn*16 + laneN;
        #pragma unroll
        for (int v = 0; v < 8; ++v) {
            int row = mbase + wm*32 + tm*16 + 8*laneG + v;
            out[(size_t)row * DM + col] = acc[tm][tn][v];
        }
      }
}

// ---------------------------------------------------------------------------
extern "C" void kernel_launch(void* const* d_in, const int* in_sizes, int n_in,
                              void* d_out, int out_size, void* d_ws, size_t ws_size,
                              hipStream_t stream)
{
    const float* x  = (const float*)d_in[0];
    const float* Wq = (const float*)d_in[1];
    const float* bq = (const float*)d_in[2];
    const float* Wk = (const float*)d_in[3];
    const float* bk = (const float*)d_in[4];
    const float* Wv = (const float*)d_in[5];
    const float* bv = (const float*)d_in[6];
    const float* Wo = (const float*)d_in[7];
    const float* bo = (const float*)d_in[8];
    float* out = (float*)d_out;

    // workspace: 4 x 64MB bf16 buffers (q[head][pos][d], k[head][pos][d],
    // v^T[head][d][pos], o[head][pos][d])
    __bf16* qh = (__bf16*)d_ws;
    __bf16* kh = qh + HEAD_BUF;
    __bf16* vT = kh + HEAD_BUF;
    __bf16* oh = vT + HEAD_BUF;

    dim3 pgrid(NQK / 256, ROWS / 64), pblk(256);
    proj_kernel<<<pgrid, pblk, 0, stream>>>(x, Wq, bq, qh, 0);
    proj_kernel<<<pgrid, pblk, 0, stream>>>(x, Wk, bk, kh, 0);
    proj_kernel<<<pgrid, pblk, 0, stream>>>(x, Wv, bv, vT, 1);

    attn_kernel<<<dim3(4, NHEADS), dim3(128), 0, stream>>>(qh, kh, vT, oh);

    outproj_kernel<<<dim3(1, ROWS / 64), dim3(256), 0, stream>>>(oh, Wo, bo, out);
}
